// MultiHead_Attention_Lattice_rel_36893769073025
// MI455X (gfx1250) — matrix operand order repaired and checked by reading, not compiled, not run on hardware
//
#include <hip/hip_runtime.h>
#include <hip/hip_bf16.h>

// Problem constants (reference: B=2, L=256, H=512, NH=8, DH=64, MAXLEN=256)
#define BB      2
#define LL      256
#define HH      512
#define NHH     8
#define DHH     64
#define MAXLEN_ 256
#define PEROWS  513   // 2*MAXLEN+1

typedef __attribute__((ext_vector_type(16))) __bf16 v16bf;
typedef __attribute__((ext_vector_type(2)))  __bf16 v2bf;
typedef __attribute__((ext_vector_type(8)))  float  v8f;

union BF16Frag {
  v16bf v;
  unsigned short s[16];
  unsigned int u[8];
  uint4 q[2];
};

__device__ __forceinline__ unsigned short f2bf(float x) {
  // round-to-nearest-even fp32 -> bf16
  unsigned int u = __float_as_uint(x);
  return (unsigned short)((u + 0x7FFFu + ((u >> 16) & 1u)) >> 16);
}
__device__ __forceinline__ unsigned int pk2(float lo, float hi) {
#if __has_builtin(__builtin_amdgcn_cvt_pk_bf16_f32)
  v2bf t = __builtin_amdgcn_cvt_pk_bf16_f32(lo, hi);
  return __builtin_bit_cast(unsigned int, t);
#else
  return (unsigned int)f2bf(lo) | ((unsigned int)f2bf(hi) << 16);
#endif
}
__device__ __forceinline__ uint4 pk8(float4 a, float4 b) {
  uint4 r;
  r.x = pk2(a.x, a.y); r.y = pk2(a.z, a.w);
  r.z = pk2(b.x, b.y); r.w = pk2(b.z, b.w);
  return r;
}

// ---------------------------------------------------------------------------
// Generic tiled WMMA GEMM: C[m,n] = sum_k A[m,k] * B(k,n) (+bias[n]) (relu?)
//   bMode 0: B(k,n) = Bm[n*ldb + k]   (torch Linear weight, K-contiguous)
//   bMode 1: B(k,n) = Bm[k*ldb + n]   (N-contiguous)
// REQUIRES: K % 32 == 0, N % 64 == 0 (true for every call below); only M ragged.
// Per-z pointer offsets: off = (z/zdiv)*s1 + (z%zdiv)*s2 for A, B, C.
// Block = 256 threads = 8 waves; block tile 64x64; wave tile 16x32 (2 WMMA N).
// LDS tiles are both [row][k] so every WMMA fragment is contiguous 16B runs
// per lane -> ds_load_b128. A-row is clamped so staging loads are always legal
// (zeroing done with selects, no exec-mask branches in the hot loop).
// ---------------------------------------------------------------------------
__global__ __launch_bounds__(256) void gemm_wmma_kernel(
    const float* __restrict__ Abase, const float* __restrict__ Bbase,
    const float* __restrict__ bias, float* __restrict__ Cbase,
    int M, int N, int K, int lda, int ldb, int ldc, int bMode,
    int zdiv, int sA1, int sA2, int sB1, int sB2, int sC1, int sC2, int doRelu)
{
  __shared__ alignas(16) unsigned short As[64][40];   // [m][k] bf16 (pad 80B rows)
  __shared__ alignas(16) unsigned short Bs[64][40];   // [n][k] bf16 (pad 80B rows)

  const int z = blockIdx.z;
  const float* A  = Abase + (size_t)(z / zdiv) * sA1 + (size_t)(z % zdiv) * sA2;
  const float* Bm = Bbase + (size_t)(z / zdiv) * sB1 + (size_t)(z % zdiv) * sB2;
  float*       C  = Cbase + (size_t)(z / zdiv) * sC1 + (size_t)(z % zdiv) * sC2;

  const int bm = blockIdx.y * 64;
  const int bn = blockIdx.x * 64;
  const int tid  = threadIdx.x;
  const int wave = tid >> 5, lane = tid & 31;
  const int wm = (wave & 3) * 16;   // wave M tile in block
  const int wn = (wave >> 2) * 32;  // wave N half in block
  const int g  = lane >> 4;         // lane half
  const int lm = lane & 15;

  const int arow = tid >> 2;                       // 0..63 (A stage row)
  const int aks  = (tid & 3) * 8;                  // 0..24 (A stage k segment)
  const bool amOk = (bm + arow) < M;
  const int arowC = min(bm + arow, M - 1);         // clamped: loads always legal
  const float* aRow = A + (size_t)arowC * lda;

  v8f acc0 = {0.f,0.f,0.f,0.f,0.f,0.f,0.f,0.f};
  v8f acc1 = {0.f,0.f,0.f,0.f,0.f,0.f,0.f,0.f};

  #pragma unroll 2
  for (int k0 = 0; k0 < K; k0 += 32) {
    // ---- stage A tile 64x32 (unconditional b128 loads + select-zero) ----
    {
      const float4* src = (const float4*)&aRow[k0 + aks];
      if (k0 + 32 < K)
        __builtin_prefetch(&aRow[k0 + 32 + aks], 0, 3);
      float4 u0 = src[0], u1 = src[1];
      if (!amOk) {
        u0 = make_float4(0.f, 0.f, 0.f, 0.f);
        u1 = make_float4(0.f, 0.f, 0.f, 0.f);
      }
      *(uint4*)&As[arow][aks] = pk8(u0, u1);
    }
    // ---- stage B tile into Bs[n][k] (no guards: N%64==0, K%32==0) ----
    if (bMode == 0) {
      int n  = tid >> 2;
      int ks = (tid & 3) * 8;
      const float4* src = (const float4*)&Bm[(size_t)(bn + n) * ldb + k0 + ks];
      float4 u0 = src[0], u1 = src[1];
      *(uint4*)&Bs[n][ks] = pk8(u0, u1);
    } else {
      int kk = tid >> 3;            // 0..31
      int ns = (tid & 7) * 8;       // 0..56
      const float4* src = (const float4*)&Bm[(size_t)(k0 + kk) * ldb + bn + ns];
      float4 u0 = src[0], u1 = src[1];
      unsigned int pkd[4];
      pkd[0] = pk2(u0.x, u0.y); pkd[1] = pk2(u0.z, u0.w);
      pkd[2] = pk2(u1.x, u1.y); pkd[3] = pk2(u1.z, u1.w);
      #pragma unroll
      for (int e = 0; e < 4; ++e) {
        Bs[ns + 2 * e][kk]     = (unsigned short)(pkd[e] & 0xFFFFu);
        Bs[ns + 2 * e + 1][kk] = (unsigned short)(pkd[e] >> 16);
      }
    }
    __syncthreads();

    // ---- fragments: contiguous 16B runs per lane -> ds_load_b128 ----
    BF16Frag af;
    af.q[0] = *(const uint4*)&As[wm + lm][g * 16];
    af.q[1] = *(const uint4*)&As[wm + lm][g * 16 + 8];
    BF16Frag bf0, bf1;
    bf0.q[0] = *(const uint4*)&Bs[wn + lm][g * 16];
    bf0.q[1] = *(const uint4*)&Bs[wn + lm][g * 16 + 8];
    bf1.q[0] = *(const uint4*)&Bs[wn + 16 + lm][g * 16];
    bf1.q[1] = *(const uint4*)&Bs[wn + 16 + lm][g * 16 + 8];

    acc0 = __builtin_amdgcn_wmma_f32_16x16x32_bf16(false, af.v, false, bf0.v,
                                                   (short)0, acc0, false, false);
    acc1 = __builtin_amdgcn_wmma_f32_16x16x32_bf16(false, af.v, false, bf1.v,
                                                   (short)0, acc1, false, false);
    __syncthreads();
  }

  // ---- epilogue: C/D layout m = r + 8*(lane/16), n = lane%16 ----
  #pragma unroll
  for (int r = 0; r < 8; ++r) {
    int gm = bm + wm + r + 8 * g;
    if (gm < M) {
      int n0g = bn + wn + lm;
      int n1g = n0g + 16;
      float v0 = acc0[r], v1 = acc1[r];
      if (bias) { v0 += bias[n0g]; v1 += bias[n1g]; }
      if (doRelu) { v0 = fmaxf(v0, 0.f); v1 = fmaxf(v1, 0.f); }
      C[(size_t)gm * ldc + n0g] = v0;
      C[(size_t)gm * ldc + n1g] = v1;
    }
  }
}

// ---------------------------------------------------------------------------
// qv[r,c] = qp[r,c] + vflat[c]      (v is (NH,DH) == flat H layout)
// ---------------------------------------------------------------------------
__global__ __launch_bounds__(256) void addv_kernel(
    const float* __restrict__ qp, const float* __restrict__ vparam,
    float* __restrict__ qv)
{
  int t = blockIdx.x * 256 + threadIdx.x;   // 0 .. B*L*H-1
  qv[t] = qp[t] + vparam[t & (HH - 1)];
}

// ---------------------------------------------------------------------------
// Cterm[b,h,j] = sum_d u[h,d] * kp[b,j,h*DH+d]
// ---------------------------------------------------------------------------
__global__ __launch_bounds__(256) void cterm_kernel(
    const float* __restrict__ kp, const float* __restrict__ u,
    float* __restrict__ Cterm)
{
  int t = blockIdx.x * 256 + threadIdx.x;  // 0 .. B*NH*L-1
  int b = t >> 11;
  int h = (t >> 8) & 7;
  int j = t & (LL - 1);
  const float4* kr = (const float4*)(kp + ((size_t)(b * LL + j)) * HH + h * DHH);
  const float4* ur = (const float4*)(u + h * DHH);
  float s = 0.f;
  #pragma unroll
  for (int d = 0; d < DHH / 4; ++d) {
    float4 a = ur[d], k4 = kr[d];
    s += a.x * k4.x + a.y * k4.y + a.z * k4.z + a.w * k4.w;
  }
  Cterm[t] = s;
}

// ---------------------------------------------------------------------------
// Flagship: rel-score contribution. One block per (b,i); 8 waves cover 16
// j-tiles. rel[b,i,j,n] = relu(P0[ss]+P1[se]+P2[es]+P3[ee]+bf)[n] built on the
// fly from L2-resident P tables; contracted against w[b,:,i,:] via WMMA
// (M=16: 8 heads zero-padded; N=16 j's; K=512 in 16 steps of 32).
// scores[b,h,i,j] += result  (A-term already written by GEMM).
// ---------------------------------------------------------------------------
__global__ __launch_bounds__(256) void relscore_kernel(
    const float* __restrict__ P,     // [4][513][512]
    const float* __restrict__ bfv,   // [512]
    const float* __restrict__ wbuf,  // [B][L][NH][H]
    const int* __restrict__ pos_s, const int* __restrict__ pos_e,
    float* __restrict__ scores)      // [B][NH][L][L]
{
  __shared__ alignas(16) unsigned short wlds[16][520];  // bf16 w rows, 8..15 zero
  const int i = blockIdx.x;
  const int b = blockIdx.y;
  const int tid  = threadIdx.x;
  const int wave = tid >> 5, lane = tid & 31;
  const int g = lane >> 4, lj = lane & 15;

  { // stage w rows (A-matrix), zero-pad heads 8..15; packed uint4 stores
    int row = tid >> 4;            // 0..15
    int seg = (tid & 15) * 32;     // 0..480 step 32
    if (row < NHH) {
      const float4* src =
          (const float4*)(wbuf + (((size_t)(b * LL + i)) * NHH + row) * HH + seg);
      #pragma unroll
      for (int e = 0; e < 4; ++e) {
        float4 u0 = src[2 * e], u1 = src[2 * e + 1];
        *(uint4*)&wlds[row][seg + e * 8] = pk8(u0, u1);
      }
    } else {
      uint4 zz = make_uint4(0u, 0u, 0u, 0u);
      #pragma unroll
      for (int e = 0; e < 4; ++e) *(uint4*)&wlds[row][seg + e * 8] = zz;
    }
  }
  __syncthreads();

  const int psi = pos_s[b * LL + i];
  const int pei = pos_e[b * LL + i];
  const float4* bv4 = (const float4*)bfv;

  for (int it = 0; it < 2; ++it) {
    const int jt = wave + it * 8;          // j-tile 0..15
    const int j  = jt * 16 + lj;
    const int jps = pos_s[b * LL + j];
    const int jpe = pos_e[b * LL + j];
    int i0 = min(max(psi - jps + MAXLEN_, 0), PEROWS - 1);
    int i1 = min(max(psi - jpe + MAXLEN_, 0), PEROWS - 1);
    int i2 = min(max(pei - jps + MAXLEN_, 0), PEROWS - 1);
    int i3 = min(max(pei - jpe + MAXLEN_, 0), PEROWS - 1);
    const float4* r0 = (const float4*)(P + (size_t)i0 * HH);
    const float4* r1 = (const float4*)(P + (size_t)PEROWS * HH + (size_t)i1 * HH);
    const float4* r2 = (const float4*)(P + 2 * (size_t)PEROWS * HH + (size_t)i2 * HH);
    const float4* r3 = (const float4*)(P + 3 * (size_t)PEROWS * HH + (size_t)i3 * HH);

    v8f acc = {0.f,0.f,0.f,0.f,0.f,0.f,0.f,0.f};
    #pragma unroll 4
    for (int kc = 0; kc < 16; ++kc) {
      const int n0 = kc * 32 + g * 16;     // this lane-half's 16 n values
      BF16Frag brf;                        // B fragment: rel[n0+e][j]
      #pragma unroll
      for (int q4 = 0; q4 < 4; ++q4) {
        int f4 = (n0 >> 2) + q4;
        float4 a  = r0[f4];
        float4 b1 = r1[f4];
        float4 c  = r2[f4];
        float4 d  = r3[f4];
        float4 e  = bv4[f4];
        float x0 = fmaxf(a.x + b1.x + c.x + d.x + e.x, 0.f);
        float x1 = fmaxf(a.y + b1.y + c.y + d.y + e.y, 0.f);
        float x2 = fmaxf(a.z + b1.z + c.z + d.z + e.z, 0.f);
        float x3 = fmaxf(a.w + b1.w + c.w + d.w + e.w, 0.f);
        brf.u[q4 * 2]     = pk2(x0, x1);
        brf.u[q4 * 2 + 1] = pk2(x2, x3);
      }
      BF16Frag afr;                        // A fragment: two contiguous 16B runs
      afr.q[0] = *(const uint4*)&wlds[lj][kc * 32 + g * 16];
      afr.q[1] = *(const uint4*)&wlds[lj][kc * 32 + g * 16 + 8];

      acc = __builtin_amdgcn_wmma_f32_16x16x32_bf16(false, afr.v, false, brf.v,
                                                    (short)0, acc, false, false);
    }
    #pragma unroll
    for (int r = 0; r < 8; ++r) {
      int m = r + 8 * g;                   // head index (rows >=8 are zero pad)
      if (m < NHH) {
        size_t idx = (((size_t)b * NHH + m) * LL + i) * LL + jt * 16 + lj;
        scores[idx] += acc[r];
      }
    }
  }
}

// ---------------------------------------------------------------------------
// Row softmax: s = (scores + Cterm[b,h,j]) * 1/sqrt(DH), mask j>=total_len,
// softmax over j. One wave per row (wave32), 8 elems/lane, shfl reductions.
// ---------------------------------------------------------------------------
__global__ __launch_bounds__(256) void softmax_kernel(
    float* __restrict__ scores, const float* __restrict__ Cterm,
    const int* __restrict__ seq_len, const int* __restrict__ lex_num)
{
  int wave = threadIdx.x >> 5, lane = threadIdx.x & 31;
  int row = blockIdx.x * 8 + wave;       // 0 .. B*NH*L-1
  int b = row >> 11;
  int h = (row >> 8) & 7;
  int tl = seq_len[b] + lex_num[b];
  float* sr = scores + (size_t)row * LL;
  const float* cr = Cterm + ((size_t)b * NHH + h) * LL;

  float x[8];
  int j0 = lane * 8;
  const float4* s4 = (const float4*)(sr + j0);
  const float4* c4 = (const float4*)(cr + j0);
  float4 sa = s4[0], sb = s4[1], ca = c4[0], cb = c4[1];
  x[0] = (sa.x + ca.x) * 0.125f; x[1] = (sa.y + ca.y) * 0.125f;
  x[2] = (sa.z + ca.z) * 0.125f; x[3] = (sa.w + ca.w) * 0.125f;
  x[4] = (sb.x + cb.x) * 0.125f; x[5] = (sb.y + cb.y) * 0.125f;
  x[6] = (sb.z + cb.z) * 0.125f; x[7] = (sb.w + cb.w) * 0.125f;
  #pragma unroll
  for (int e = 0; e < 8; ++e)
    if (j0 + e >= tl) x[e] = -1e15f;

  float mx = x[0];
  #pragma unroll
  for (int e = 1; e < 8; ++e) mx = fmaxf(mx, x[e]);
  for (int o = 16; o > 0; o >>= 1) mx = fmaxf(mx, __shfl_xor(mx, o));
  float sum = 0.f;
  #pragma unroll
  for (int e = 0; e < 8; ++e) { x[e] = __expf(x[e] - mx); sum += x[e]; }
  for (int o = 16; o > 0; o >>= 1) sum += __shfl_xor(sum, o);
  float inv = 1.f / sum;
  float4 o0 = make_float4(x[0] * inv, x[1] * inv, x[2] * inv, x[3] * inv);
  float4 o1 = make_float4(x[4] * inv, x[5] * inv, x[6] * inv, x[7] * inv);
  ((float4*)(sr + j0))[0] = o0;
  ((float4*)(sr + j0))[1] = o1;
}

// ---------------------------------------------------------------------------
extern "C" void kernel_launch(void* const* d_in, const int* in_sizes, int n_in,
                              void* d_out, int out_size, void* d_ws, size_t ws_size,
                              hipStream_t stream) {
  const float* key    = (const float*)d_in[0];
  const float* query  = (const float*)d_in[1];
  const float* value  = (const float*)d_in[2];
  const float* pe     = (const float*)d_in[3];
  const float* Wf     = (const float*)d_in[4];
  const float* bfv    = (const float*)d_in[5];
  const float* Wk     = (const float*)d_in[6];
  const float* bk     = (const float*)d_in[7];
  const float* Wq     = (const float*)d_in[8];
  const float* bq     = (const float*)d_in[9];
  const float* Wv     = (const float*)d_in[10];
  const float* bv     = (const float*)d_in[11];
  const float* Wr     = (const float*)d_in[12];
  // d_in[13] = br: j-independent contribution -> softmax-invariant -> dropped
  const float* Wo     = (const float*)d_in[14];
  const float* bo     = (const float*)d_in[15];
  const float* u      = (const float*)d_in[16];
  const float* vparam = (const float*)d_in[17];
  const int* seq_len  = (const int*)d_in[18];
  const int* lex_num  = (const int*)d_in[19];
  const int* pos_s    = (const int*)d_in[20];
  const int* pos_e    = (const int*)d_in[21];

  const int BLH = BB * LL * HH;            // 262144
  float* ws = (float*)d_ws;
  size_t off = 0;
  float* P        = ws + off; off += (size_t)4 * PEROWS * HH;   // 1,050,624
  float* qp       = ws + off; off += BLH;
  float* kp       = ws + off; off += BLH;
  float* vp       = ws + off; off += BLH;
  float* qv       = ws + off; off += BLH;
  float* wbuf     = ws + off; off += (size_t)BB * LL * NHH * HH; // 2,097,152
  float* Cterm    = ws + off; off += (size_t)BB * NHH * LL;      // 4,096
  float* scores   = ws + off; off += (size_t)BB * NHH * LL * LL; // 1,048,576
  float* attn_out = ws + off; off += BLH;
  // total ~22.0 MB of the preallocated workspace

  // 1) P_c = pe @ Wf_c.T  (4 x [513x512], K=512) — folds gather through Wf
  gemm_wmma_kernel<<<dim3(8, 9, 4), 256, 0, stream>>>(
      pe, Wf, nullptr, P, PEROWS, HH, HH, HH, 4 * HH, HH,
      /*bMode*/0, /*zdiv*/4, 0, 0, 0, HH, 0, PEROWS * HH, 0);

  // 2) q/k/v projections: [B*L,H] @ [H,H].T + bias
  gemm_wmma_kernel<<<dim3(8, 8, 1), 256, 0, stream>>>(
      query, Wq, bq, qp, BB * LL, HH, HH, HH, HH, HH, 0, 1, 0,0,0,0,0,0, 0);
  gemm_wmma_kernel<<<dim3(8, 8, 1), 256, 0, stream>>>(
      key, Wk, bk, kp, BB * LL, HH, HH, HH, HH, HH, 0, 1, 0,0,0,0,0,0, 0);
  gemm_wmma_kernel<<<dim3(8, 8, 1), 256, 0, stream>>>(
      value, Wv, bv, vp, BB * LL, HH, HH, HH, HH, HH, 0, 1, 0,0,0,0,0,0, 0);

  // 3) qv = qp + v (broadcast)
  addv_kernel<<<BLH / 256, 256, 0, stream>>>(qp, vparam, qv);

  // 4) w[b,i,h,:] = qv[b,i,h*DH:] @ Wr[h*DH:(h+1)*DH, :]  (z = head, K=64)
  gemm_wmma_kernel<<<dim3(8, 8, NHH), 256, 0, stream>>>(
      qv, Wr, nullptr, wbuf, BB * LL, HH, DHH, HH, HH, NHH * HH,
      /*bMode*/1, /*zdiv*/NHH, 0, DHH, 0, DHH * HH, 0, HH, 0);

  // 5) Cterm[b,h,j] = u[h] . k[b,h,j]
  cterm_kernel<<<(BB * NHH * LL) / 256, 256, 0, stream>>>(kp, u, Cterm);

  // 6) A-term: scores[b,h,:,:] = q_h @ k_h.T  (z = b*NH+h, K=64)
  gemm_wmma_kernel<<<dim3(4, 4, BB * NHH), 256, 0, stream>>>(
      qp, kp, nullptr, scores, LL, LL, DHH, HH, HH, LL,
      /*bMode*/0, /*zdiv*/NHH, LL * HH, DHH, LL * HH, DHH,
      NHH * LL * LL, LL * LL, 0);

  // 7) rel contribution: scores += w . relu(P gathers + bf)
  relscore_kernel<<<dim3(LL, BB), 256, 0, stream>>>(
      P, bfv, wbuf, pos_s, pos_e, scores);

  // 8) masked, scaled softmax (+ j-dependent C term)
  softmax_kernel<<<(BB * NHH * LL) / 8, 256, 0, stream>>>(
      scores, Cterm, seq_len, lex_num);

  // 9) out_h = attn_h @ v_h  (z = b*NH+h, [256x256]x[256x64])
  gemm_wmma_kernel<<<dim3(1, 4, BB * NHH), 256, 0, stream>>>(
      scores, vp, nullptr, attn_out, LL, DHH, LL, LL, HH, HH,
      /*bMode*/1, /*zdiv*/NHH, NHH * LL * LL, LL * LL, LL * HH, DHH,
      LL * HH, DHH, 0);

  // 10) final projection: d_out = attn_out @ Wo.T + bo
  gemm_wmma_kernel<<<dim3(8, 8, 1), 256, 0, stream>>>(
      attn_out, Wo, bo, (float*)d_out, BB * LL, HH, HH, HH, HH, HH,
      0, 1, 0,0,0,0,0,0, 0);

  (void)in_sizes; (void)n_in; (void)out_size; (void)ws_size;
}
